// TopKSAE_11828339933558
// MI455X (gfx1250) — matrix operand order repaired and checked by reading, not compiled, not run on hardware
//
#include <hip/hip_runtime.h>
#include <hip/hip_bf16.h>

typedef __attribute__((ext_vector_type(16))) _Float16 v16h;
typedef __attribute__((ext_vector_type(8)))  _Float16 v8h;
typedef __attribute__((ext_vector_type(4)))  _Float16 v4h;
typedef __attribute__((ext_vector_type(8)))  float    v8f;

#define D_IN_   1024
#define D_SAE_  16384
#define BATCH_  8192
#define TOPK_   32

#define TILE_M 128
#define TILE_N 256
#define TILE_K 32
#define LDS_PAD 8
#define AB_STRIDE (TILE_K + LDS_PAD)   // 40 halves = 80 bytes (multiple of 16)

// ---------------------------------------------------------------------------
// Kernel 1: pre_acts = relu(x @ W_enc + b_enc), f16 WMMA, f32 accumulate.
// Block = 256 threads (8 wave32). Block tile 128x256, K-step 32.
// Waves in a 2(M) x 4(N) grid; each wave owns a 64x64 sub-tile =
// 4x4 WMMA accumulators -> 16 v_wmma per K-step, A/B frags each reused 4x.
// Next-tile global_prefetch_b8 issued unconditionally (speculative, SE scope:
// OOB translation failures on the final tile are silently dropped per ISA).
// ---------------------------------------------------------------------------
__global__ __launch_bounds__(256) void sae_enc_gemm(
    const float* __restrict__ X,      // [BATCH, D_IN]
    const float* __restrict__ W,      // [D_IN, D_SAE]
    const float* __restrict__ benc,   // [D_SAE]
    float* __restrict__ Pre)          // [BATCH, D_SAE]
{
  __shared__ alignas(16) _Float16 As[TILE_M * AB_STRIDE];   // [m][k] 10 KB
  __shared__ alignas(16) _Float16 Bt[TILE_N * AB_STRIDE];   // [n][k] 20 KB (transposed)

  const int tid   = threadIdx.x;
  const int lane  = tid & 31;
  const int wv    = tid >> 5;
  const int waveM = wv >> 2;          // 0..1  -> 64-row slice
  const int waveN = wv & 3;           // 0..3  -> 64-col slice
  const int mBase = blockIdx.y * TILE_M;
  const int nBase = blockIdx.x * TILE_N;

  const int lm   = lane & 15;         // row/col within 16
  const int kgrp = lane >> 4;         // 0 or 1 (lane half)
  const int akb  = kgrp * 8;          // A frag K sub-base (ISA 16-bit A layout)
  const int bkb  = kgrp * 8;          // B frag K sub-base (same layout as A)

  v8f acc[4][4];
  v8f zero = {};
#pragma unroll
  for (int mi = 0; mi < 4; ++mi)
#pragma unroll
    for (int ni = 0; ni < 4; ++ni) acc[mi][ni] = zero;

  for (int k0 = 0; k0 < D_IN_; k0 += TILE_K) {
    // --- load A tile (128x32 f32 -> f16), 4 float4 per thread, coalesced ---
#pragma unroll
    for (int i = 0; i < 4; ++i) {
      int e = tid + i * 256;          // float4 id, 1024 total
      int r = e >> 3;                 // 0..127
      int c = (e & 7) << 2;           // 0,4,...,28
      const float* gp = X + (size_t)(mBase + r) * D_IN_ + k0 + c;
      const float4 f = *(const float4*)gp;
      __builtin_prefetch(gp + TILE_K, 0, 3);   // speculative next-tile prefetch
      v4h h; h[0] = (_Float16)f.x; h[1] = (_Float16)f.y;
             h[2] = (_Float16)f.z; h[3] = (_Float16)f.w;
      *(v4h*)&As[r * AB_STRIDE + c] = h;
    }
    // --- load B tile (32x256 f32), store transposed [n][k] as f16 ---
#pragma unroll
    for (int i = 0; i < 8; ++i) {
      int e = tid + i * 256;          // float4 id, 2048 total
      int r = e >> 6;                 // k row 0..31
      int c = (e & 63) << 2;          // n col 0,4,...,252
      const float* gp = W + (size_t)(k0 + r) * D_SAE_ + nBase + c;
      const float4 f = *(const float4*)gp;
      __builtin_prefetch(gp + (size_t)TILE_K * D_SAE_, 0, 3);
      Bt[(c + 0) * AB_STRIDE + r] = (_Float16)f.x;
      Bt[(c + 1) * AB_STRIDE + r] = (_Float16)f.y;
      Bt[(c + 2) * AB_STRIDE + r] = (_Float16)f.z;
      Bt[(c + 3) * AB_STRIDE + r] = (_Float16)f.w;
    }
    __syncthreads();

    // --- B fragments (32x16 f16): lane holds col n, K=bkb..bkb+7, 16+bkb..16+bkb+7 ---
    v16h bf[4];
#pragma unroll
    for (int ni = 0; ni < 4; ++ni) {
      int bn = waveN * 64 + ni * 16 + lm;
      v8h blo = *(const v8h*)&Bt[bn * AB_STRIDE + bkb];
      v8h bhi = *(const v8h*)&Bt[bn * AB_STRIDE + 16 + bkb];
      bf[ni] = __builtin_shufflevector(blo, bhi,
               0,1,2,3,4,5,6,7,8,9,10,11,12,13,14,15);
    }

    // --- A fragments (16x32 f16) + 16 WMMAs ---
#pragma unroll
    for (int mi = 0; mi < 4; ++mi) {
      int am = waveM * 64 + mi * 16 + lm;
      v8h alo = *(const v8h*)&As[am * AB_STRIDE + akb];        // K = akb..akb+7
      v8h ahi = *(const v8h*)&As[am * AB_STRIDE + 16 + akb];   // K = 16+akb..
      v16h af = __builtin_shufflevector(alo, ahi,
                0,1,2,3,4,5,6,7,8,9,10,11,12,13,14,15);
#pragma unroll
      for (int ni = 0; ni < 4; ++ni)
        acc[mi][ni] = __builtin_amdgcn_wmma_f32_16x16x32_f16(
            false, af, false, bf[ni], (short)0, acc[mi][ni], false, false);
    }
    __syncthreads();
  }

  // --- epilogue: bias + relu, C layout: lane=col, VGPR r => row r (+8 for hi lanes)
#pragma unroll
  for (int mi = 0; mi < 4; ++mi)
#pragma unroll
    for (int ni = 0; ni < 4; ++ni) {
      int col = nBase + waveN * 64 + ni * 16 + lm;
      float bb = benc[col];
#pragma unroll
      for (int r = 0; r < 8; ++r) {
        int row = mBase + waveM * 64 + mi * 16 + kgrp * 8 + r;
        float vv = acc[mi][ni][r] + bb;
        Pre[(size_t)row * D_SAE_ + col] = vv > 0.f ? vv : 0.f;
      }
    }
}

// ---------------------------------------------------------------------------
// Kernel 2: per-row top-k via uniform binary search on float-as-uint threshold
// (values >= 0 after relu, so uint order == float order). Deterministic.
// ---------------------------------------------------------------------------
__global__ __launch_bounds__(256) void sae_topk(
    float* __restrict__ H,          // [BATCH, D_SAE] pre_acts in, sparse out
    float* __restrict__ pair_val,   // [BATCH*TOPK]
    int*   __restrict__ pair_idx,   // [BATCH*TOPK]
    float* __restrict__ l0_part)    // [BATCH]
{
  const int row = blockIdx.x;
  const int tid = threadIdx.x;
  float* hrow = H + (size_t)row * D_SAE_;

  float v[64];
#pragma unroll
  for (int i = 0; i < 64; ++i) v[i] = hrow[tid + i * 256];

  __shared__ int red[256];

  unsigned lo = 0u, hi = 0x7F800000u;   // [0, +inf)
  while (hi - lo > 1u) {
    unsigned mid = (lo + hi) >> 1;
    float mf = __uint_as_float(mid);
    int c = 0;
#pragma unroll
    for (int i = 0; i < 64; ++i) c += (v[i] > mf) ? 1 : 0;
    red[tid] = c;
    __syncthreads();
    for (int s = 128; s > 0; s >>= 1) {
      if (tid < s) red[tid] += red[tid + s];
      __syncthreads();
    }
    int total = red[0];
    __syncthreads();
    if (total >= TOPK_) lo = mid; else hi = mid;
  }
  const float thr = __uint_as_float(lo);

  // deterministic exclusive scan of per-thread keep counts (orders pairs by tid)
  int myc = 0;
#pragma unroll
  for (int i = 0; i < 64; ++i) myc += (v[i] > thr) ? 1 : 0;
  red[tid] = myc;
  __syncthreads();
  for (int off = 1; off < 256; off <<= 1) {
    int t = (tid >= off) ? red[tid - off] : 0;
    __syncthreads();
    red[tid] += t;
    __syncthreads();
  }
  int base  = red[tid] - myc;
  int total = red[255];

  int s = base;
#pragma unroll
  for (int i = 0; i < 64; ++i) {
    int idx = tid + i * 256;
    float val = v[i];
    bool keep = val > thr;
    hrow[idx] = keep ? val : 0.f;
    if (keep) {
      if (s < TOPK_) { pair_val[row * TOPK_ + s] = val; pair_idx[row * TOPK_ + s] = idx; }
      ++s;
    }
  }
  if (tid == 0) {
    for (int t = total; t < TOPK_; ++t) { pair_val[row * TOPK_ + t] = 0.f; pair_idx[row * TOPK_ + t] = 0; }
    l0_part[row] = (float)total;
  }
}

// ---------------------------------------------------------------------------
// Kernel 3: x_hat = sum_j val_j * W_dec[idx_j,:] + b_dec ; per-row sq-err partial
// ---------------------------------------------------------------------------
__global__ __launch_bounds__(256) void sae_decode(
    const float* __restrict__ X,
    const float* __restrict__ Wdec,      // [D_SAE, D_IN]
    const float* __restrict__ bdec,      // [D_IN]
    const float* __restrict__ pair_val,
    const int*   __restrict__ pair_idx,
    float* __restrict__ Xhat,            // [BATCH, D_IN]
    float* __restrict__ loss_part)       // [BATCH]
{
  const int row = blockIdx.x;
  const int tid = threadIdx.x;
  __shared__ float pv[TOPK_];
  __shared__ int   pi[TOPK_];
  if (tid < TOPK_) { pv[tid] = pair_val[row * TOPK_ + tid]; pi[tid] = pair_idx[row * TOPK_ + tid]; }
  __syncthreads();

  const int d = tid * 4;
  float4 b4 = *(const float4*)(bdec + d);
  float ax = b4.x, ay = b4.y, az = b4.z, aw = b4.w;
  for (int j = 0; j < TOPK_; ++j) {
    float w = pv[j];
    if (w == 0.f) continue;          // uniform across block
    const float4 wd = *(const float4*)(Wdec + (size_t)pi[j] * D_IN_ + d);
    ax += w * wd.x; ay += w * wd.y; az += w * wd.z; aw += w * wd.w;
  }
  float4 out; out.x = ax; out.y = ay; out.z = az; out.w = aw;
  *(float4*)(Xhat + (size_t)row * D_IN_ + d) = out;

  const float4 xr = *(const float4*)(X + (size_t)row * D_IN_ + d);
  float ex = ax - xr.x, ey = ay - xr.y, ez = az - xr.z, ew = aw - xr.w;
  float sq = ex * ex + ey * ey + ez * ez + ew * ew;

  __shared__ float red[256];
  red[tid] = sq;
  __syncthreads();
  for (int s = 128; s > 0; s >>= 1) {
    if (tid < s) red[tid] += red[tid + s];
    __syncthreads();
  }
  if (tid == 0) loss_part[row] = red[0];
}

// ---------------------------------------------------------------------------
// Kernel 4: deterministic final reduction -> recon_loss, l0
// ---------------------------------------------------------------------------
__global__ __launch_bounds__(256) void sae_finalize(
    const float* __restrict__ loss_part,
    const float* __restrict__ l0_part,
    float* __restrict__ scalars)         // [0]=recon_loss, [1]=l0
{
  const int tid = threadIdx.x;
  float s1 = 0.f, s2 = 0.f;
  for (int i = tid; i < BATCH_; i += 256) { s1 += loss_part[i]; s2 += l0_part[i]; }
  __shared__ float r1[256], r2[256];
  r1[tid] = s1; r2[tid] = s2;
  __syncthreads();
  for (int s = 128; s > 0; s >>= 1) {
    if (tid < s) { r1[tid] += r1[tid + s]; r2[tid] += r2[tid + s]; }
    __syncthreads();
  }
  if (tid == 0) {
    scalars[0] = r1[0] / (float)((size_t)BATCH_ * D_IN_);
    scalars[1] = r2[0] / (float)BATCH_;
  }
}

// ---------------------------------------------------------------------------
extern "C" void kernel_launch(void* const* d_in, const int* in_sizes, int n_in,
                              void* d_out, int out_size, void* d_ws, size_t ws_size,
                              hipStream_t stream) {
  (void)in_sizes; (void)n_in; (void)out_size; (void)ws_size;
  const float* X    = (const float*)d_in[0];
  const float* Wenc = (const float*)d_in[1];
  const float* benc = (const float*)d_in[2];
  const float* Wdec = (const float*)d_in[3];
  const float* bdec = (const float*)d_in[4];

  float* out = (float*)d_out;
  const size_t XHAT_ELEMS = (size_t)BATCH_ * D_IN_;     // 8388608
  const size_t H_ELEMS    = (size_t)BATCH_ * D_SAE_;    // 134217728
  float* xhat    = out;
  float* hsparse = out + XHAT_ELEMS;
  float* scalars = out + XHAT_ELEMS + H_ELEMS;

  char* ws = (char*)d_ws;
  float* pair_val  = (float*)ws;                                       // 1 MB
  int*   pair_idx  = (int*)(ws + (size_t)BATCH_ * TOPK_ * 4);          // 1 MB
  float* loss_part = (float*)(ws + (size_t)BATCH_ * TOPK_ * 8);        // 32 KB
  float* l0_part   = loss_part + BATCH_;                               // 32 KB

  dim3 gridGemm(D_SAE_ / TILE_N, BATCH_ / TILE_M);     // (64, 64)
  sae_enc_gemm<<<gridGemm, 256, 0, stream>>>(X, Wenc, benc, hsparse);
  sae_topk<<<BATCH_, 256, 0, stream>>>(hsparse, pair_val, pair_idx, l0_part);
  sae_decode<<<BATCH_, 256, 0, stream>>>(X, Wdec, bdec, pair_val, pair_idx, xhat, loss_part);
  sae_finalize<<<1, 256, 0, stream>>>(loss_part, l0_part, scalars);
}
